// MoChA_88596585382101
// MI455X (gfx1250) — matrix operand order repaired and checked by reading, not compiled, not run on hardware
//
#include <hip/hip_runtime.h>
#include <hip/hip_bf16.h>
#include <math.h>

#define B_    16
#define KMAX_ 4096
#define D_    512
#define A_    512
#define EPS_  1e-10f

typedef __bf16 bf16_t;
typedef __attribute__((ext_vector_type(16))) __bf16 v16bf;
typedef __attribute__((ext_vector_type(8)))  __bf16 v8bf;
typedef __attribute__((ext_vector_type(4)))  __bf16 v4bf;
typedef __attribute__((ext_vector_type(8)))  float  v8f;
typedef __attribute__((ext_vector_type(4)))  float  f32x4;

// ---------------------------------------------------------------- prep: v_w
__global__ __launch_bounds__(512) void prep_vw(const float* __restrict__ v_v,
                                               const float* __restrict__ v_g,
                                               float* __restrict__ vw) {
    __shared__ float ssum;
    const int t = threadIdx.x;
    if (t == 0) ssum = 0.f;
    __syncthreads();
    float v = v_v[t];
    float s = v * v;
    #pragma unroll
    for (int m = 16; m >= 1; m >>= 1) s += __shfl_xor(s, m, 32);
    if ((t & 31) == 0) atomicAdd(&ssum, s);
    __syncthreads();
    vw[t] = v_g[0] * v / sqrtf(ssum);
}

// ------------------------------------------------------- prep: wk_w -> bf16
__global__ __launch_bounds__(256) void prep_wkw(const float* __restrict__ w,
                                                bf16_t* __restrict__ o) {
    int i = blockIdx.x * 256 + threadIdx.x;      // 65536 threads
    #pragma unroll
    for (int r = 0; r < 4; ++r) o[i + r * 65536] = (bf16_t)w[i + r * 65536];
}

// --------------------------------------------- prep: qb = q@wq_w^T + wk_b
__global__ __launch_bounds__(512) void prep_qb(const float* __restrict__ query,
                                               const float* __restrict__ wq_w,
                                               const float* __restrict__ wk_b,
                                               float* __restrict__ qb) {
    const int b = blockIdx.x, a = threadIdx.x;
    const float* q = query + b * D_;
    const float* w = wq_w + (size_t)a * D_;
    float s = 0.f;
    for (int d = 0; d < D_; ++d) s += q[d] * w[d];
    qb[b * A_ + a] = s + wk_b[a];
}

// --------------------- fused projection GEMM + tanh-dot epilogue (WMMA bf16)
__global__ __launch_bounds__(128) void mocha_proj(
    const float*  __restrict__ key_enc, const bf16_t* __restrict__ wkw,
    const float*  __restrict__ qb,      const float*  __restrict__ vw,
    const float*  __restrict__ r,       float* __restrict__ e_out) {
    __shared__ __align__(16) bf16_t ldsA[4 * 16 * D_];   // 64 KB
    const int wave = threadIdx.x >> 5;
    const int lane = threadIdx.x & 31;
    const int b    = blockIdx.x >> 6;                    // 64 row-groups/batch
    const int rg   = blockIdx.x & 63;
    const int waveRow = rg * 64 + wave * 16;

    // Stage this wave's 16x512 fp32 key rows into LDS as bf16 (coalesced).
    const float* src = key_enc + ((size_t)b * KMAX_ + waveRow) * D_;
    bf16_t* myLds = ldsA + wave * (16 * D_);
    #pragma unroll 4
    for (int c = 0; c < 64; ++c) {
        int i = (c * 32 + lane) * 4;
        f32x4 f = *(const f32x4*)(src + i);
        v4bf h;
        h[0] = (bf16_t)f[0]; h[1] = (bf16_t)f[1];
        h[2] = (bf16_t)f[2]; h[3] = (bf16_t)f[3];
        *(v4bf*)(myLds + i) = h;
    }
    __syncthreads();

    // Load all 16 A fragments (K-chunks of 32) once; contiguous K order (same as B): lane 16h+m
    // holds row m, K = {16h..16h+15}  -> two b128 LDS loads.
    const int m = lane & 15, h = lane >> 4;
    const bf16_t* rowp = myLds + m * D_;
    v16bf afrag[16];
    #pragma unroll
    for (int kc = 0; kc < 16; ++kc) {
        v8bf lo = *(const v8bf*)(rowp + kc * 32 + h * 16);
        v8bf hi = *(const v8bf*)(rowp + kc * 32 + h * 16 + 8);
        afrag[kc] = __builtin_shufflevector(lo, hi,
            0,1,2,3,4,5,6,7,8,9,10,11,12,13,14,15);
    }

    const float r0 = r[0];
    float ep[8] = {0,0,0,0,0,0,0,0};

    for (int n = 0; n < 32; ++n) {                       // N tiles of 16
        const int aGlob = n * 16 + m;                    // column for B lanes
        const bf16_t* bp = wkw + (size_t)aGlob * D_ + h * 16;
        v8f cacc = {};
        #pragma unroll
        for (int kc = 0; kc < 16; ++kc) {                // K chunks of 32
            v16bf bfrag = *(const v16bf*)(bp + kc * 32); // 16 contig bf16/lane
            cacc = __builtin_amdgcn_wmma_f32_16x16x32_bf16(
                false, afrag[kc], false, bfrag, (short)0, cacc, false, false);
        }
        // Epilogue: lane 16h+n holds rows v+8h of column aGlob.
        const float qv = qb[b * A_ + aGlob];
        const float wv = vw[aGlob];
        #pragma unroll
        for (int v = 0; v < 8; ++v)
            ep[v] += tanhf(cacc[v] + qv) * wv;
    }

    // Reduce the 16 column-lanes of each half-wave (masks stay within halves).
    #pragma unroll
    for (int v = 0; v < 8; ++v) {
        float s = ep[v];
        s += __shfl_xor(s, 1, 32); s += __shfl_xor(s, 2, 32);
        s += __shfl_xor(s, 4, 32); s += __shfl_xor(s, 8, 32);
        ep[v] = s;
    }
    if (m == 0) {                                        // lanes 0 and 16
        #pragma unroll
        for (int v = 0; v < 8; ++v)
            e_out[(size_t)b * KMAX_ + waveRow + h * 8 + v] = ep[v] + r0;
    }
}

// ---------------------- monotonic-attention scan: aw = p_choose * cp (fused)
__global__ __launch_bounds__(256) void mocha_scan(
    const float* __restrict__ e_mono, const float* __restrict__ noise,
    float* __restrict__ out) {
    const int b = blockIdx.x;
    const int t = threadIdx.x;
    // re-zero the cv accumulator region every call (output is poisoned once)
    out[b * D_ + t]       = 0.f;
    out[b * D_ + 256 + t] = 0.f;

    const float* eb = e_mono + (size_t)b * KMAX_;
    const float* nb = noise  + (size_t)b * KMAX_;
    float larr[16];   // s[16t+i]: prepended 1 at j==0, else log1mp[j-1]
    float parr[16];   // p_choose[16t+i]
    #pragma unroll
    for (int i2 = 0; i2 <= 16; ++i2) {
        int idx = t * 16 - 1 + i2;
        float p = 0.f, l = 1.f;                  // repo quirk: s[0] = 1
        if (idx >= 0) {
            float x = eb[idx] + nb[idx];
            p = 1.f / (1.f + __expf(-x));
            l = __logf(fmaxf(1.f - p, EPS_));
        }
        if (i2 < 16) larr[i2] = l;
        if (i2 >= 1) parr[i2 - 1] = p;
    }
    float run[16], acc = 0.f;
    #pragma unroll
    for (int i = 0; i < 16; ++i) { acc += larr[i]; run[i] = acc; }

    __shared__ float sdata[256];
    sdata[t] = acc;
    __syncthreads();
    const float mine = acc;
    for (int off = 1; off < 256; off <<= 1) {   // Hillis-Steele inclusive scan
        float x = (t >= off) ? sdata[t - off] : 0.f;
        __syncthreads();
        sdata[t] += x;
        __syncthreads();
    }
    const float excl = sdata[t] - mine;

    float* aw = out + B_ * D_ + (size_t)b * KMAX_;
    #pragma unroll
    for (int i = 0; i < 16; ++i) {
        float cp = __expf(excl + run[i]);
        aw[t * 16 + i] = parr[i] * cp;          // cumsum(aw_prev/clip(cp)) == 1
    }
}

// -------------------------------------------- cv = aw @ value (memory-bound)
__global__ __launch_bounds__(128) void mocha_cv(const float* __restrict__ value,
                                                float* __restrict__ out) {
    const int b  = blockIdx.x >> 3;
    const int sl = blockIdx.x & 7;
    const int t  = threadIdx.x;
    const float* aw = out + B_ * D_ + (size_t)b * KMAX_;
    f32x4 acc = {0.f, 0.f, 0.f, 0.f};
    const int kBeg = sl * (KMAX_ / 8);
    for (int k = kBeg; k < kBeg + KMAX_ / 8; ++k) {
        float a = aw[k];
        f32x4 v = *(const f32x4*)(value + ((size_t)b * KMAX_ + k) * D_ + t * 4);
        acc[0] += a * v[0]; acc[1] += a * v[1];
        acc[2] += a * v[2]; acc[3] += a * v[3];
    }
    float* cv = out + b * D_ + t * 4;
    atomicAdd(cv + 0, acc[0]);
    atomicAdd(cv + 1, acc[1]);
    atomicAdd(cv + 2, acc[2]);
    atomicAdd(cv + 3, acc[3]);
}

extern "C" void kernel_launch(void* const* d_in, const int* in_sizes, int n_in,
                              void* d_out, int out_size, void* d_ws, size_t ws_size,
                              hipStream_t stream) {
    const float* key_enc = (const float*)d_in[0];
    const float* value   = (const float*)d_in[1];
    const float* query   = (const float*)d_in[2];
    const float* noise   = (const float*)d_in[3];
    const float* wk_w    = (const float*)d_in[4];
    const float* wk_b    = (const float*)d_in[5];
    const float* wq_w    = (const float*)d_in[6];
    const float* v_v     = (const float*)d_in[7];
    const float* v_g     = (const float*)d_in[8];
    const float* r       = (const float*)d_in[9];
    float* out = (float*)d_out;

    char* ws = (char*)d_ws;
    bf16_t* wkw_bf = (bf16_t*)ws;                              // 512 KB
    float*  qb     = (float*)(ws + 524288);                    //  32 KB
    float*  vw     = (float*)(ws + 524288 + 32768);            //   2 KB
    float*  e_mono = (float*)(ws + 524288 + 32768 + 2048);     // 256 KB

    prep_vw  <<<1,        512, 0, stream>>>(v_v, v_g, vw);
    prep_wkw <<<256,      256, 0, stream>>>(wk_w, wkw_bf);
    prep_qb  <<<B_,       512, 0, stream>>>(query, wq_w, wk_b, qb);
    mocha_proj<<<B_ * 64, 128, 0, stream>>>(key_enc, wkw_bf, qb, vw, r, e_mono);
    mocha_scan<<<B_,      256, 0, stream>>>(e_mono, noise, out);
    mocha_cv  <<<B_ * 8,  128, 0, stream>>>(value, out);
}
